// SimpleGCN_81870666596916
// MI455X (gfx1250) — matrix operand order repaired and checked by reading, not compiled, not run on hardware
//
#include <hip/hip_runtime.h>

typedef __attribute__((ext_vector_type(16))) _Float16 v16h;
typedef __attribute__((ext_vector_type(8)))  float    v8f;

#define N_FEAT 64   // input/hidden feature width (A-matrix row stride everywhere)

// ---------------------------------------------------------------------------
// Pre-convert a weight matrix W[64 x FOUT] (f32, row-major) into f16 laid out
// in WMMA B-fragment order so each lane loads its whole fragment contiguously:
//   Wf[((colTile*2 + kc)*32 + lane)*16 + i] = (f16) W[(kc*32 + (i<8 ? i : i+8) + 8*(lane/16))*FOUT
//                                                     + colTile*16 + lane%16]
// Total elements = FOUT*64 (same count as W). Runs once per launch; tiny.
// ---------------------------------------------------------------------------
template<int FOUT>
__global__ __launch_bounds__(256) void k_convw(const float* __restrict__ W,
                                               _Float16* __restrict__ Wf) {
  int t = blockIdx.x * blockDim.x + threadIdx.x;
  if (t >= FOUT * 64) return;
  const int i       = t & 15;
  const int lane    = (t >> 4) & 31;
  const int kc      = (t >> 9) & 1;
  const int colTile = t >> 10;
  const int g  = lane >> 4;
  const int n  = colTile * 16 + (lane & 15);
  const int k  = kc * 32 + 8 * g + (i < 8 ? i : i + 8);
  Wf[t] = (_Float16)W[k * FOUT + n];
}

// ---------------------------------------------------------------------------
// WMMA GEMM:  C[nrows x FOUT] = A[nrows x 64] @ W[64 x FOUT]  (+bias, +relu)
// One wave32 per 16x16 output tile; K=64 via two 16x16x32 f16 WMMAs.
// Fragment layouts per CDNA5 ISA 7.12.2 (wave32):
//   A 16x32 f16 : lane L holds row M=L%16; g=L/16 selects K-half;
//                 elem i -> K = (i<8 ? i+8g : i+8+8g) -> two contiguous 8-runs.
//   B 32x16 f16 : pre-packed fragment, contiguous 32 B per lane (see k_convw).
//   D 16x16 f32 : VGPR r of lane L -> (M = r + 8g, N = L%16).
// ---------------------------------------------------------------------------
template<int FOUT, bool RELU, bool BIAS>
__global__ __launch_bounds__(256) void k_gemm_wmma(
    const float* __restrict__ A, const _Float16* __restrict__ Wf,
    const float* __restrict__ bias, float* __restrict__ C, int nrows)
{
  constexpr int COLT = FOUT / 16;
  const int wave = threadIdx.x >> 5;
  const int lane = threadIdx.x & 31;
  const int tile = blockIdx.x * 8 + wave;            // 8 waves / 256-thread block
  const int rowTile = tile / COLT;
  const int colTile = tile - rowTile * COLT;
  const int row0 = rowTile << 4;
  if (row0 >= nrows) return;                          // whole-wave uniform exit

  const int l15 = lane & 15;
  const int g   = lane >> 4;
  const int n   = (colTile << 4) + l15;
  const float* __restrict__ arow = A + (size_t)(row0 + l15) * N_FEAT;

  v8f acc = {};
#pragma unroll
  for (int kc = 0; kc < 2; ++kc) {
    const int kb = kc * 32;

    // A fragment: two contiguous 8-float runs -> 4x global_load_b128
    const float4 a0 = *(const float4*)(arow + kb + 8 * g);
    const float4 a1 = *(const float4*)(arow + kb + 8 * g + 4);
    const float4 a2 = *(const float4*)(arow + kb + 16 + 8 * g);
    const float4 a3 = *(const float4*)(arow + kb + 16 + 8 * g + 4);
    v16h a;
    a[0]  = (_Float16)a0.x; a[1]  = (_Float16)a0.y;
    a[2]  = (_Float16)a0.z; a[3]  = (_Float16)a0.w;
    a[4]  = (_Float16)a1.x; a[5]  = (_Float16)a1.y;
    a[6]  = (_Float16)a1.z; a[7]  = (_Float16)a1.w;
    a[8]  = (_Float16)a2.x; a[9]  = (_Float16)a2.y;
    a[10] = (_Float16)a2.z; a[11] = (_Float16)a2.w;
    a[12] = (_Float16)a3.x; a[13] = (_Float16)a3.y;
    a[14] = (_Float16)a3.z; a[15] = (_Float16)a3.w;

    // B fragment: contiguous 32 B per lane -> 2x global_load_b128
    const v16h b = *(const v16h*)(Wf + ((size_t)(colTile * 2 + kc) * 32 + lane) * 16);

    // (neg_a, A, neg_b, B, c_mod, C, reuse_a, reuse_b)
    acc = __builtin_amdgcn_wmma_f32_16x16x32_f16(false, a, false, b,
                                                 (short)0, acc, false, false);
  }

#pragma unroll
  for (int r = 0; r < 8; ++r) {
    const int m = row0 + r + 8 * g;
    float v = acc[r];
    if (BIAS) v += bias[n];
    if (RELU) v = fmaxf(v, 0.0f);
    C[(size_t)m * FOUT + n] = v;
  }
}

// ---------------------------------------------------------------------------
// Elementwise helpers
// ---------------------------------------------------------------------------
__global__ __launch_bounds__(256) void k_fill(float* __restrict__ p, float v, int n) {
  int t = blockIdx.x * blockDim.x + threadIdx.x;
  if (t < n) p[t] = v;
}

__global__ __launch_bounds__(256) void k_deg_count(const int* __restrict__ dst,
                                                   float* __restrict__ deg, int E) {
  int t = blockIdx.x * blockDim.x + threadIdx.x;
  if (t < E) atomicAdd(&deg[dst[t]], 1.0f);
}

__global__ __launch_bounds__(256) void k_rsqrt_inplace(float* __restrict__ p, int n) {
  int t = blockIdx.x * blockDim.x + threadIdx.x;
  if (t < n) p[t] = rsqrtf(p[t]);   // deg >= 1 always (self-loop)
}

// Edge scatter: out[dst] += xw[src] * dinv[src]*dinv[dst].
// 16 threads per edge, one float4 (16 B) each -> 64 floats per edge.
__global__ __launch_bounds__(256) void k_scatter(
    const float* __restrict__ xw, const int* __restrict__ src,
    const int* __restrict__ dst, const float* __restrict__ dinv,
    float* __restrict__ out, int E)
{
  int t = blockIdx.x * blockDim.x + threadIdx.x;
  if (t >= E * 16) return;
  const int e = t >> 4;
  const int q = t & 15;
  const int s = src[e];
  const int d = dst[e];
  const float nrm = dinv[s] * dinv[d];
  const float4 v = ((const float4*)(xw + (size_t)s * N_FEAT))[q];
  float* o = out + (size_t)d * N_FEAT + (q << 2);
  atomicAdd(o + 0, v.x * nrm);
  atomicAdd(o + 1, v.y * nrm);
  atomicAdd(o + 2, v.z * nrm);
  atomicAdd(o + 3, v.w * nrm);
}

// Fold self-loop contribution + bias + relu (no atomics; one thread per element)
__global__ __launch_bounds__(256) void k_finalize(
    const float* __restrict__ xw, const float* __restrict__ dinv,
    const float* __restrict__ bias, float* __restrict__ h, int total)
{
  int t = blockIdx.x * blockDim.x + threadIdx.x;
  if (t >= total) return;
  const int node = t >> 6;
  const int f    = t & 63;
  const float di = dinv[node];
  const float v  = h[t] + xw[t] * (di * di) + bias[f];
  h[t] = fmaxf(v, 0.0f);
}

// ---------------------------------------------------------------------------
extern "C" void kernel_launch(void* const* d_in, const int* in_sizes, int n_in,
                              void* d_out, int out_size, void* d_ws, size_t ws_size,
                              hipStream_t stream) {
  const float* x   = (const float*)d_in[0];
  const int*   ei  = (const int*)  d_in[1];
  const float* W1  = (const float*)d_in[2];
  const float* b1  = (const float*)d_in[3];
  const float* W2  = (const float*)d_in[4];
  const float* b2  = (const float*)d_in[5];
  const float* dW1 = (const float*)d_in[6];
  const float* db1 = (const float*)d_in[7];
  const float* dW2 = (const float*)d_in[8];
  const float* db2 = (const float*)d_in[9];
  float* out = (float*)d_out;

  const int N = in_sizes[0] / N_FEAT;     // 100000
  const int E = in_sizes[1] / 2;          // 1600000
  const int* src = ei;                    // edge_index row 0
  const int* dst = ei + E;                // edge_index row 1

  float* bufA = (float*)d_ws;                       // xw / h3      (N*64 f32)
  float* bufB = bufA + (size_t)N * N_FEAT;          // agg/h1/h2    (N*64 f32)
  float* dinv = bufB + (size_t)N * N_FEAT;          // deg -> dinv  (N f32)
  _Float16* wf1 = (_Float16*)(dinv + N);            // W1  fragments (64*64 f16)
  _Float16* wf2 = wf1 + 64 * 64;                    // W2  fragments
  _Float16* wf3 = wf2 + 64 * 64;                    // dW1 fragments
  _Float16* wf4 = wf3 + 64 * 64;                    // dW2 fragments (64*16 f16)

  const int NF = N * N_FEAT;
  auto cdiv = [](int a, int b) { return (a + b - 1) / b; };
  dim3 blk(256);

  const int g64 = cdiv(cdiv(N, 16) * 4, 8);   // 16x16 tiles, 4 col-tiles, 8 waves/blk
  const int g16 = cdiv(cdiv(N, 16) * 1, 8);

  // --- pack weights into f16 WMMA B-fragments (tiny, once per launch) ---
  k_convw<64><<<cdiv(64 * 64, 256), blk, 0, stream>>>(W1,  wf1);
  k_convw<64><<<cdiv(64 * 64, 256), blk, 0, stream>>>(W2,  wf2);
  k_convw<64><<<cdiv(64 * 64, 256), blk, 0, stream>>>(dW1, wf3);
  k_convw<16><<<cdiv(16 * 64, 256), blk, 0, stream>>>(dW2, wf4);

  // --- degree normalization: dinv = rsqrt(indeg + 1) ---
  k_fill<<<cdiv(N, 256), blk, 0, stream>>>(dinv, 1.0f, N);
  k_deg_count<<<cdiv(E, 256), blk, 0, stream>>>(dst, dinv, E);
  k_rsqrt_inplace<<<cdiv(N, 256), blk, 0, stream>>>(dinv, N);

  // --- GCN layer 1 ---
  k_gemm_wmma<64, false, false><<<g64, blk, 0, stream>>>(x, wf1, nullptr, bufA, N);
  k_fill<<<cdiv(NF, 256), blk, 0, stream>>>(bufB, 0.0f, NF);
  k_scatter<<<cdiv(E * 16, 256), blk, 0, stream>>>(bufA, src, dst, dinv, bufB, E);
  k_finalize<<<cdiv(NF, 256), blk, 0, stream>>>(bufA, dinv, b1, bufB, NF);   // h1 in bufB

  // --- GCN layer 2 ---
  k_gemm_wmma<64, false, false><<<g64, blk, 0, stream>>>(bufB, wf2, nullptr, bufA, N);
  k_fill<<<cdiv(NF, 256), blk, 0, stream>>>(bufB, 0.0f, NF);
  k_scatter<<<cdiv(E * 16, 256), blk, 0, stream>>>(bufA, src, dst, dinv, bufB, E);
  k_finalize<<<cdiv(NF, 256), blk, 0, stream>>>(bufA, dinv, b2, bufB, NF);   // h2 in bufB

  // --- MLP head ---
  k_gemm_wmma<64, true,  true ><<<g64, blk, 0, stream>>>(bufB, wf3, db1, bufA, N); // h3
  k_gemm_wmma<16, false, true ><<<g16, blk, 0, stream>>>(bufA, wf4, db2, out,  N); // out
}
